// BinaryLinearLayer_44607530336600
// MI455X (gfx1250) — 8-bit matrix operand order repaired and checked by reading, compiled, not run on hardware
//
#include <hip/hip_runtime.h>
#include <stdint.h>


#define MDIM 8192
#define KDIM 4096
#define NDIM 4096

#define TPAD 144
#define ATILE_BYTES (128 * TPAD)
#define BUF_BYTES   (2 * ATILE_BYTES)

typedef __attribute__((ext_vector_type(16))) int   v16i;
typedef __attribute__((ext_vector_type(8)))  float v8f;

__device__ __forceinline__ uint8_t sign_fp8(float x) {
    return x > 0.0f ? (uint8_t)0x38 : (x < 0.0f ? (uint8_t)0xB8 : (uint8_t)0x00);
}

__device__ __forceinline__ size_t kswizzle(size_t i) {
    const size_t g = (i >> 3) & 3;
    const size_t gmap = ((g & 1) << 1) | (g >> 1);
    return (i & ~(size_t)31) | (gmap << 3) | (i & 7);
}

__device__ __forceinline__ void wait_async0() {
#if __has_builtin(__builtin_amdgcn_s_wait_asynccnt)
    __builtin_amdgcn_s_wait_asynccnt(0);
#else
    asm volatile("s_wait_asynccnt 0x0" ::: "memory");
#endif
}

__device__ __forceinline__ void async_b128(uint32_t lds_off, const uint8_t* g) {
    asm volatile("global_load_async_to_lds_b128 %0, %1, off"
                 :: "v"(lds_off), "v"((unsigned long long)(uintptr_t)g)
                 : "memory");
}

__global__ void __launch_bounds__(256) quant_x_kernel(const float* __restrict__ x,
                                                      uint8_t* __restrict__ x8s) {
    const size_t i = ((size_t)blockIdx.x * blockDim.x + threadIdx.x) * 4;
    const float4 v = *(const float4*)(x + i);
    uchar4 o;
    o.x = sign_fp8(v.x);
    o.y = sign_fp8(v.y);
    o.z = sign_fp8(v.z);
    o.w = sign_fp8(v.w);
    *(uchar4*)(x8s + i) = o;
}

__global__ void __launch_bounds__(256) quant_wT_kernel(const float* __restrict__ w,
                                                       uint8_t* __restrict__ w8t) {
    __shared__ uint8_t tile[64][64 + 4];

    const int n0 = blockIdx.x * 64;
    const int k0 = blockIdx.y * 64;
    const int tc = threadIdx.x & 63;
    const int tr = threadIdx.x >> 6;

#pragma unroll
    for (int i = 0; i < 16; ++i) {
        const int k = tr + i * 4;
        tile[k][tc] = sign_fp8(w[(size_t)(k0 + k) * NDIM + (n0 + tc)]);
    }
    __syncthreads();
#pragma unroll
    for (int i = 0; i < 16; ++i) {
        const int n = tr + i * 4;
        w8t[(size_t)(n0 + n) * KDIM + (k0 + tc)] = tile[tc][n];
    }
}

union Frag {
    v16i  v;
    uint4 q[4];
};

__global__ void __launch_bounds__(256) bnn_gemm_kernel(const uint8_t* __restrict__ a8s,
                                                       const uint8_t* __restrict__ b8t,
                                                       float* __restrict__ out) {
    __shared__ __align__(16) uint8_t smem[2 * BUF_BYTES];

    const int t     = threadIdx.x;
    const int lane  = t & 31;
    const int wave  = t >> 5;
    const int lrow  = lane & 15;
    const int lhalf = lane >> 4;

    const int wm = (wave & 1) * 64;
    const int wn = (wave >> 1) * 32;

    const uint8_t* a_tile_g = a8s + (size_t)(blockIdx.x * 128) * KDIM;
    const uint8_t* b_tile_g = b8t + (size_t)(blockIdx.y * 128) * KDIM;

    const uint32_t smem_base = (uint32_t)(uintptr_t)&smem[0];

    const int crow = t >> 3;
    const int coff = (t & 7) << 4;

    auto issue_tiles = [&](int buf, int kb) {
#pragma unroll
        for (int p = 0; p < 4; ++p) {
            const int r = crow + 32 * p;
            async_b128(smem_base + buf * BUF_BYTES + r * TPAD + coff,
                       a_tile_g + (size_t)r * KDIM + kb + coff);
        }
#pragma unroll
        for (int p = 0; p < 4; ++p) {
            const int r = crow + 32 * p;
            async_b128(smem_base + buf * BUF_BYTES + ATILE_BYTES + r * TPAD + coff,
                       b_tile_g + (size_t)r * KDIM + kb + coff);
        }
    };

    v8f acc[4][2];
#pragma unroll
    for (int f = 0; f < 4; ++f)
#pragma unroll
        for (int g = 0; g < 2; ++g)
            acc[f][g] = (v8f)(0.0f);

    issue_tiles(0, 0);

    for (int kb = 0; kb < KDIM; kb += 128) {
        const int cur = (kb >> 7) & 1;

        wait_async0();
        __syncthreads();

        if (kb + 128 < KDIM)
            issue_tiles(cur ^ 1, kb + 128);

        const uint8_t* als = &smem[cur * BUF_BYTES];
        const uint8_t* bls = &smem[cur * BUF_BYTES + ATILE_BYTES];

        Frag bf[2];
#pragma unroll
        for (int g = 0; g < 2; ++g)
#pragma unroll
            for (int c = 0; c < 4; ++c)
                bf[g].q[c] = *(const uint4*)(bls + (wn + 16 * g + lrow) * TPAD +
                                             (lhalf << 4) + 32 * c);
#pragma unroll
        for (int f = 0; f < 4; ++f) {
            Frag af;
#pragma unroll
            for (int c = 0; c < 4; ++c)
                af.q[c] = *(const uint4*)(als + (wm + 16 * f + lrow) * TPAD +
                                          (lhalf << 4) + 32 * c);
            acc[f][0] = __builtin_amdgcn_wmma_f32_16x16x128_fp8_fp8(
                            af.v, bf[0].v, (short)0, acc[f][0], false, false);
            acc[f][1] = __builtin_amdgcn_wmma_f32_16x16x128_fp8_fp8(
                            af.v, bf[1].v, (short)0, acc[f][1], false, false);
        }

        __syncthreads();
    }

    const int m0 = blockIdx.x * 128 + wm;
    const int n0 = blockIdx.y * 128 + wn;
#pragma unroll
    for (int f = 0; f < 4; ++f)
#pragma unroll
        for (int g = 0; g < 2; ++g) {
            const int n  = n0 + 16 * g + lrow;
            const int mb = m0 + 16 * f + (lhalf << 3);
#pragma unroll
            for (int r = 0; r < 8; ++r)
                out[(size_t)(mb + r) * NDIM + n] = acc[f][g][r];
        }
}

extern "C" void kernel_launch(void* const* d_in, const int* in_sizes, int n_in,
                              void* d_out, int out_size, void* d_ws, size_t ws_size,
                              hipStream_t stream) {
    (void)in_sizes; (void)n_in; (void)out_size; (void)ws_size;

    const float* x = (const float*)d_in[0];
    const float* w = (const float*)d_in[1];
    float* out = (float*)d_out;

    uint8_t* x8s = (uint8_t*)d_ws;
    uint8_t* w8t = (uint8_t*)d_ws + (size_t)MDIM * KDIM;

    {
        const size_t threads = ((size_t)MDIM * KDIM) / 4;
        quant_x_kernel<<<dim3((unsigned)(threads / 256)), dim3(256), 0, stream>>>(x, x8s);
    }
    quant_wT_kernel<<<dim3(NDIM / 64, KDIM / 64), dim3(256), 0, stream>>>(w, w8t);
    bnn_gemm_kernel<<<dim3(MDIM / 128, NDIM / 128), dim3(256), 0, stream>>>(x8s, w8t, out);
}
